// FusionNetLite_46557445489145
// MI455X (gfx1250) — compile-verified, matrix operand order checked by reading, not run on hardware
//
#include <hip/hip_runtime.h>
#include <stdint.h>

#define S_DIRS 7
#define KNN_K  10
#define BATCH  4
#define NP1    2048
#define NP2    512
#define NP3    128

#define CEILDIV(a,b) (((a)+(b)-1)/(b))

typedef __attribute__((ext_vector_type(16))) __bf16 v16bf;
typedef __attribute__((ext_vector_type(8)))  float  v8f;

__device__ __forceinline__ unsigned short f32_to_bf16(float f) {
  unsigned int u = __float_as_uint(f);
  u += 0x7FFFu + ((u >> 16) & 1u);   // round-to-nearest-even
  return (unsigned short)(u >> 16);
}

// ---------------- direction matrix column-normalize: out[:,m] = d[:,m]/max(||.||,1e-12)
__global__ void k_norm_dirs(const float* __restrict__ d, float* __restrict__ o, int D, int M) {
  int m = blockIdx.x * blockDim.x + threadIdx.x;
  if (m >= M) return;
  float s = 0.f;
  for (int i = 0; i < D; ++i) { float v = d[(long)i * M + m]; s += v * v; }
  float inv = 1.f / fmaxf(sqrtf(s), 1e-12f);
  for (int i = 0; i < D; ++i) o[(long)i * M + m] = d[(long)i * M + m] * inv;
}

// ---------------- brute-force KNN: out[b,i,k] = index of (k+1)-th nearest (skip self)
__global__ void k_knn(const float* __restrict__ pts, int N, int D, int K1,
                      int* __restrict__ out, int Kout) {
  int b = blockIdx.y;
  int i = blockIdx.x * blockDim.x + threadIdx.x;
  __shared__ float tile[128 * 9];
  float q[9];
  float bd[11]; int bi[11];
  bool act = (i < N);
  if (act) for (int d = 0; d < D; ++d) q[d] = pts[((long)b * N + i) * D + d];
  for (int k = 0; k < K1; ++k) { bd[k] = 3.4e38f; bi[k] = 0; }
  for (int j0 = 0; j0 < N; j0 += 128) {
    int nt = min(128, N - j0);
    __syncthreads();
    for (int t = threadIdx.x; t < nt * D; t += blockDim.x)
      tile[t] = pts[((long)b * N + j0) * D + t];
    __syncthreads();
    if (act) {
      for (int j = 0; j < nt; ++j) {
        float dist = 0.f;
        for (int d = 0; d < D; ++d) { float df = q[d] - tile[j * D + d]; dist += df * df; }
        if (dist < bd[K1 - 1]) {
          int p = K1 - 1;
          while (p > 0 && bd[p - 1] > dist) { bd[p] = bd[p - 1]; bi[p] = bi[p - 1]; --p; }
          bd[p] = dist; bi[p] = j0 + j;
        }
      }
    }
  }
  if (act) for (int k = 0; k < Kout; ++k) out[((long)b * N + i) * Kout + k] = bi[k + 1];
}

// ---------------- nearest source index per target point
__global__ void k_nearest(const float* __restrict__ tgt, const float* __restrict__ src,
                          int Nt, int Ns, int D, int* __restrict__ out) {
  int b = blockIdx.y;
  int i = blockIdx.x * blockDim.x + threadIdx.x;
  __shared__ float tile[128 * 9];
  float q[9];
  bool act = (i < Nt);
  if (act) for (int d = 0; d < D; ++d) q[d] = tgt[((long)b * Nt + i) * D + d];
  float best = 3.4e38f; int bidx = 0;
  for (int j0 = 0; j0 < Ns; j0 += 128) {
    int nt = min(128, Ns - j0);
    __syncthreads();
    for (int t = threadIdx.x; t < nt * D; t += blockDim.x)
      tile[t] = src[((long)b * Ns + j0) * D + t];
    __syncthreads();
    if (act) {
      for (int j = 0; j < nt; ++j) {
        float dist = 0.f;
        for (int d = 0; d < D; ++d) { float df = q[d] - tile[j * D + d]; dist += df * df; }
        if (dist < best) { best = dist; bidx = j0 + j; }
      }
    }
  }
  if (act) out[(long)b * Nt + i] = bidx;
}

// ---------------- conv_surface: out[b,v,c] = relu( sum_s max_n relu(<dir_n, Dn[:,s*OC+c]>) )
__global__ void k_conv_surface(const int* __restrict__ idx, const float* __restrict__ pts,
                               const float* __restrict__ Dn, float* __restrict__ out,
                               int V, int D, int K, int OC) {
  int bv = blockIdx.x;
  int b = bv / V;
  __shared__ float sdir[KNN_K * 9];
  int c = threadIdx.x;
  if (c < K) {
    int j = idx[(long)bv * K + c];
    float diff[9]; float s = 0.f;
    for (int d = 0; d < D; ++d) {
      float df = pts[((long)b * V + j) * D + d] - pts[(long)bv * D + d];
      diff[d] = df; s += df * df;
    }
    float inv = 1.f / fmaxf(sqrtf(s), 1e-12f);
    for (int d = 0; d < D; ++d) sdir[c * D + d] = diff[d] * inv;
  }
  __syncthreads();
  int M = S_DIRS * OC;
  float acc = 0.f;
  for (int s = 0; s < S_DIRS; ++s) {
    float dv[9];
    for (int d = 0; d < D; ++d) dv[d] = Dn[(long)d * M + s * OC + c];
    float mx = 0.f;
    for (int n = 0; n < K; ++n) {
      float dot = 0.f;
      for (int d = 0; d < D; ++d) dot += dv[d] * sdir[n * D + d];
      mx = fmaxf(mx, fmaxf(dot, 0.f));
    }
    acc += mx;
  }
  out[(long)bv * OC + c] = fmaxf(acc, 0.f);
}

// ---------------- conv_layer epilogue: center + sum_s max_n( relu(<dir,Dn>) * supp )
__global__ void k_conv_act(const int* __restrict__ idx, const float* __restrict__ pts,
                           const float* __restrict__ Dn, const float* __restrict__ fout,
                           float* __restrict__ out, int V, int D, int K, int OC) {
  int bv = blockIdx.x;
  int b = bv / V;
  __shared__ float sdir[KNN_K * 9];
  __shared__ int sj[KNN_K];
  int c = threadIdx.x;
  if (c < K) {
    int j = idx[(long)bv * K + c];
    sj[c] = j;
    float diff[9]; float s = 0.f;
    for (int d = 0; d < D; ++d) {
      float df = pts[((long)b * V + j) * D + d] - pts[(long)bv * D + d];
      diff[d] = df; s += df * df;
    }
    float inv = 1.f / fmaxf(sqrtf(s), 1e-12f);
    for (int d = 0; d < D; ++d) sdir[c * D + d] = diff[d] * inv;
  }
  __syncthreads();
  int M = S_DIRS * OC;
  int NF = (S_DIRS + 1) * OC;
  float acc = 0.f;
  for (int s = 0; s < S_DIRS; ++s) {
    float dv[9];
    for (int d = 0; d < D; ++d) dv[d] = Dn[(long)d * M + s * OC + c];
    float mx = -3.4e38f;
    for (int n = 0; n < K; ++n) {
      float dot = 0.f;
      for (int d = 0; d < D; ++d) dot += dv[d] * sdir[n * D + d];
      float th = fmaxf(dot, 0.f);
      float supp = fout[((long)b * V + sj[n]) * NF + OC + s * OC + c];
      mx = fmaxf(mx, th * supp);
    }
    acc += mx;
  }
  out[(long)bv * OC + c] = fout[(long)bv * NF + c] + acc;
}

// ---------------- batchnorm: stats then apply+relu
__global__ void k_bn_stats(const float* __restrict__ x, int NR, int C, float* __restrict__ mv) {
  int c = blockIdx.x;
  float s = 0.f, s2 = 0.f;
  for (int r = threadIdx.x; r < NR; r += blockDim.x) {
    float v = x[(long)r * C + c]; s += v; s2 += v * v;
  }
  __shared__ float sh[256], sh2[256];
  sh[threadIdx.x] = s; sh2[threadIdx.x] = s2;
  __syncthreads();
  for (int st = blockDim.x >> 1; st > 0; st >>= 1) {
    if ((int)threadIdx.x < st) { sh[threadIdx.x] += sh[threadIdx.x + st]; sh2[threadIdx.x] += sh2[threadIdx.x + st]; }
    __syncthreads();
  }
  if (threadIdx.x == 0) {
    float m = sh[0] / (float)NR;
    mv[c] = m;
    mv[C + c] = sh2[0] / (float)NR - m * m;
  }
}

__global__ void k_bn_relu(const float* __restrict__ x, const float* __restrict__ mv,
                          const float* __restrict__ g, const float* __restrict__ be,
                          float* __restrict__ y, long n, int C) {
  long i = (long)blockIdx.x * blockDim.x + threadIdx.x;
  if (i >= n) return;
  int c = (int)(i % C);
  float v = (x[i] - mv[c]) * rsqrtf(mv[C + c] + 1e-5f) * g[c] + be[c];
  y[i] = fmaxf(v, 0.f);
}

// ---------------- pool: max over K neighbors, first PN rows only
__global__ void k_pool(const float* __restrict__ feat, const int* __restrict__ idxp,
                       float* __restrict__ out, int V, int PN, int C, int K) {
  long i = (long)blockIdx.x * blockDim.x + threadIdx.x;
  long total = (long)BATCH * PN * C;
  if (i >= total) return;
  int c = (int)(i % C); long t = i / C; int p = (int)(t % PN); int b = (int)(t / PN);
  float mx = -3.4e38f;
  for (int k = 0; k < K; ++k) {
    int j = idxp[((long)b * V + p) * K + k];
    mx = fmaxf(mx, feat[((long)b * V + j) * C + c]);
  }
  out[i] = mx;
}

// ---------------- slice first Nkeep rows per batch
__global__ void k_slice(const float* __restrict__ src, float* __restrict__ dst,
                        int Nsrc, int Nkeep, int C) {
  long i = (long)blockIdx.x * blockDim.x + threadIdx.x;
  long total = (long)BATCH * Nkeep * C;
  if (i >= total) return;
  int c = (int)(i % C); long t = i / C; int p = (int)(t % Nkeep); int b = (int)(t / Nkeep);
  dst[i] = src[((long)b * Nsrc + p) * C + c];
}

// ---------------- channel concat of three (B,V,C) -> (B,V,3C)
__global__ void k_concat3(const float* __restrict__ a, const float* __restrict__ b_,
                          const float* __restrict__ c_, float* __restrict__ o, int V, int C) {
  long i = (long)blockIdx.x * blockDim.x + threadIdx.x;
  long total = (long)BATCH * V * 3 * C;
  if (i >= total) return;
  int ch = (int)(i % (3 * C)); long t = i / (3 * C);
  const float* src = (ch < C) ? a : ((ch < 2 * C) ? b_ : c_);
  o[i] = src[t * C + (ch % C)];
}

// ---------------- f32 -> bf16 (row-major, in-place layout)
__global__ void k_f2bf(const float* __restrict__ a, unsigned short* __restrict__ o, long n) {
  long i = (long)blockIdx.x * blockDim.x + threadIdx.x;
  if (i < n) o[i] = f32_to_bf16(a[i]);
}

// ---------------- weights (K,N) f32 -> (N,K) bf16 transpose
__global__ void k_w2bf_t(const float* __restrict__ w, unsigned short* __restrict__ o, int K, int N) {
  long i = (long)blockIdx.x * blockDim.x + threadIdx.x;
  if (i >= (long)K * N) return;
  int k = (int)(i % K); int n = (int)(i / K);
  o[i] = f32_to_bf16(w[(long)k * N + n]);
}

// ---------------- WMMA GEMM: C[M,N] = A[M,K](bf16) * Bt[N,K]^T(bf16) + bias[N]
// one wave per 16x16 tile; M%16==0, N%16==0, K%32==0
__global__ void k_gemm_wmma(const unsigned short* __restrict__ A,
                            const unsigned short* __restrict__ Bt,
                            const float* __restrict__ bias, float* __restrict__ C,
                            int M, int N, int K) {
  int wid  = (int)((blockIdx.x * blockDim.x + threadIdx.x) >> 5);
  int lane = threadIdx.x & 31;
  int ntiles = N >> 4;
  int total  = (M >> 4) * ntiles;
  if (wid >= total) return;                     // wave-uniform: EXEC stays all-ones for WMMA
  int mt = wid / ntiles, nt = wid % ntiles;
  int half = lane >> 4;                         // 0: K 0..7,16..23 | 1: K 8..15,24..31
  int l15  = lane & 15;
  const unsigned int* ap = (const unsigned int*)(A  + (long)(mt * 16 + l15) * K + half * 8);
  const unsigned int* bp = (const unsigned int*)(Bt + (long)(nt * 16 + l15) * K + half * 8);
  v8f acc = {0.f, 0.f, 0.f, 0.f, 0.f, 0.f, 0.f, 0.f};
  for (int k0 = 0; k0 < K; k0 += 32) {
    union { v16bf v; unsigned int u[8]; } a, b;
    const unsigned int* ak = ap + (k0 >> 1);
    const unsigned int* bk = bp + (k0 >> 1);
#pragma unroll
    for (int t = 0; t < 4; ++t) {
      a.u[t] = ak[t];  a.u[t + 4] = ak[t + 8];   // K pairs {kb+2t} and {kb+16+2t}
      b.u[t] = bk[t];  b.u[t + 4] = bk[t + 8];
    }
    acc = __builtin_amdgcn_wmma_f32_16x16x32_bf16(false, a.v, false, b.v,
                                                  (short)0, acc, false, false);
  }
  int row0 = mt * 16 + half * 8;                // VGPR r -> M = r (+8 for upper lanes)
  int col  = nt * 16 + l15;
  float bv = bias ? bias[col] : 0.f;
#pragma unroll
  for (int r = 0; r < 8; ++r)
    C[(long)(row0 + r) * N + col] = acc[r] + bv;
}

// ---------------- final upsample + concat: out (B,2048,1280)
__global__ void k_final(const float* __restrict__ fm5, const float* __restrict__ feat1,
                        const float* __restrict__ feat2, const int* __restrict__ ni1,
                        const int* __restrict__ ni2, float* __restrict__ out) {
  long i = (long)blockIdx.x * blockDim.x + threadIdx.x;
  long total = (long)BATCH * NP1 * 1280;
  if (i >= total) return;
  int ch = (int)(i % 1280); long t = i / 1280; int v = (int)(t % NP1); int b = (int)(t / NP1);
  float val;
  if (ch < 512) {
    int j = ni2[(long)b * NP1 + v];
    val = fm5[((long)b * NP3 + j) * 512 + ch];
  } else if (ch < 896) {
    int j = ni1[(long)b * NP1 + v];
    val = feat1[((long)b * NP1 + j) * 384 + (ch - 512)];
  } else {
    int j = ni1[(long)b * NP1 + v];
    val = feat2[((long)b * NP2 + j) * 384 + (ch - 896)];
  }
  out[i] = val;
}

// =====================================================================
extern "C" void kernel_launch(void* const* d_in, const int* in_sizes, int n_in,
                              void* d_out, int out_size, void* d_ws, size_t ws_size,
                              hipStream_t stream) {
  (void)in_sizes; (void)n_in; (void)out_size; (void)ws_size;
  const float* Vts = (const float*)d_in[0];
  const float* Xyz = (const float*)d_in[1];
  const float* Nml = (const float*)d_in[2];
  const float* branch_pts[3] = { Vts, Xyz, Nml };
  auto P = [&](int i) { return (const float*)d_in[i]; };
  // params: per-branch base = 3 + 11*br: d0,w1,b1,d1,g1,be1,w2,b2,d2,g2,be2 ; then w4,b4,d4,w5,b5,d5
  const int IW4 = 36, IB4 = 37, ID4 = 38, IW5 = 39, IB5 = 40, ID5 = 41;

  size_t off = 0;
  auto alloc = [&](size_t bytes) -> char* {
    off = (off + 255) & ~(size_t)255;
    char* p = (char*)d_ws + off;
    off += bytes;
    return p;
  };

  int* idx1  = (int*)alloc((size_t)BATCH * NP1 * KNN_K * 4);
  int* idxp  = (int*)alloc((size_t)BATCH * NP1 * 4 * 4);
  int* idx2  = (int*)alloc((size_t)BATCH * NP2 * KNN_K * 4);
  int* idxp2 = (int*)alloc((size_t)BATCH * NP2 * 4 * 4);
  int* idx3  = (int*)alloc((size_t)BATCH * NP3 * KNN_K * 4);
  int* ni1   = (int*)alloc((size_t)BATCH * NP1 * 4);
  int* ni2   = (int*)alloc((size_t)BATCH * NP1 * 4);
  float *dn0[3], *dn1[3], *dn2[3];
  for (int br = 0; br < 3; ++br) {
    dn0[br] = (float*)alloc((size_t)3 * 896 * 4);
    dn1[br] = (float*)alloc((size_t)3 * 896 * 4);
    dn2[br] = (float*)alloc((size_t)3 * 896 * 4);
  }
  float* dn4 = (float*)alloc((size_t)9 * 3584 * 4);
  float* dn5 = (float*)alloc((size_t)9 * 3584 * 4);
  float* featfeat = (float*)alloc((size_t)BATCH * NP1 * 9 * 4);
  float* fm0      = (float*)alloc((size_t)BATCH * NP1 * 128 * 4);
  float* convout  = (float*)alloc((size_t)BATCH * NP1 * 128 * 4);
  float* fm1[3]; for (int i = 0; i < 3; ++i) fm1[i] = (float*)alloc((size_t)BATCH * NP1 * 128 * 4);
  float* feat1 = (float*)alloc((size_t)BATCH * NP1 * 384 * 4);
  float* pp[3]; for (int i = 0; i < 3; ++i) pp[i] = (float*)alloc((size_t)BATCH * NP2 * 3 * 4);
  float* p1 = (float*)alloc((size_t)BATCH * NP2 * 9 * 4);
  float* fp1[3]; for (int i = 0; i < 3; ++i) fp1[i] = (float*)alloc((size_t)BATCH * NP2 * 128 * 4);
  float* fm2[3]; for (int i = 0; i < 3; ++i) fm2[i] = (float*)alloc((size_t)BATCH * NP2 * 128 * 4);
  float* feat2 = (float*)alloc((size_t)BATCH * NP2 * 384 * 4);
  float* fp2   = (float*)alloc((size_t)BATCH * NP3 * 384 * 4);
  float* p2    = (float*)alloc((size_t)BATCH * NP3 * 9 * 4);
  float* vp2   = (float*)alloc((size_t)BATCH * NP3 * 3 * 4);
  float* fm4   = (float*)alloc((size_t)BATCH * NP3 * 512 * 4);
  float* fm5   = (float*)alloc((size_t)BATCH * NP3 * 512 * 4);
  float* fout  = (float*)alloc((size_t)BATCH * NP1 * 1024 * 4);     // covers 512x4096 too
  unsigned short* abf = (unsigned short*)alloc((size_t)BATCH * NP1 * 128 * 2);
  unsigned short* wbf = (unsigned short*)alloc((size_t)4096 * 512 * 2);
  float* mv = (float*)alloc(256 * 4);

  auto gemm = [&](const float* Af, const float* Wf, const float* bias, float* Cout,
                  int M, int K, int Nn) {
    long an = (long)M * K;
    k_f2bf<<<CEILDIV(an, 256), 256, 0, stream>>>(Af, abf, an);
    long wn = (long)K * Nn;
    k_w2bf_t<<<CEILDIV(wn, 256), 256, 0, stream>>>(Wf, wbf, K, Nn);
    long thr = (long)(M / 16) * (Nn / 16) * 32;
    k_gemm_wmma<<<CEILDIV(thr, 256), 256, 0, stream>>>(abf, wbf, bias, Cout, M, Nn, K);
  };
  auto bnrelu = [&](const float* x, const float* g, const float* be, float* y, int NR, int Cc) {
    k_bn_stats<<<Cc, 256, 0, stream>>>(x, NR, Cc, mv);
    long n = (long)NR * Cc;
    k_bn_relu<<<CEILDIV(n, 256), 256, 0, stream>>>(x, mv, g, be, y, n, Cc);
  };

  // ---- normalized direction matrices
  for (int br = 0; br < 3; ++br) {
    int base = 3 + 11 * br;
    k_norm_dirs<<<CEILDIV(896, 128), 128, 0, stream>>>(P(base + 0), dn0[br], 3, 896);
    k_norm_dirs<<<CEILDIV(896, 128), 128, 0, stream>>>(P(base + 3), dn1[br], 3, 896);
    k_norm_dirs<<<CEILDIV(896, 128), 128, 0, stream>>>(P(base + 8), dn2[br], 3, 896);
  }
  k_norm_dirs<<<CEILDIV(3584, 128), 128, 0, stream>>>(P(ID4), dn4, 9, 3584);
  k_norm_dirs<<<CEILDIV(3584, 128), 128, 0, stream>>>(P(ID5), dn5, 9, 3584);

  // ---- level-1 KNN and feature-space concat
  k_knn<<<dim3(CEILDIV(NP1, 128), BATCH), 128, 0, stream>>>(Vts, NP1, 3, KNN_K + 1, idx1, KNN_K);
  {
    long n = (long)BATCH * NP1 * 9;
    k_concat3<<<CEILDIV(n, 256), 256, 0, stream>>>(Vts, Xyz, Nml, featfeat, NP1, 3);
  }

  // ---- layer0 + layer1 per branch
  for (int br = 0; br < 3; ++br) {
    int base = 3 + 11 * br;
    const float* pts = branch_pts[br];
    k_conv_surface<<<BATCH * NP1, 128, 0, stream>>>(idx1, pts, dn0[br], fm0, NP1, 3, KNN_K, 128);
    gemm(fm0, P(base + 1), P(base + 2), fout, BATCH * NP1, 128, 1024);
    k_conv_act<<<BATCH * NP1, 128, 0, stream>>>(idx1, pts, dn1[br], fout, convout, NP1, 3, KNN_K, 128);
    bnrelu(convout, P(base + 4), P(base + 5), fm1[br], BATCH * NP1, 128);
  }
  {
    long n = (long)BATCH * NP1 * 384;
    k_concat3<<<CEILDIV(n, 256), 256, 0, stream>>>(fm1[0], fm1[1], fm1[2], feat1, NP1, 128);
  }

  // ---- level-1 pooling (per-branch); fp1 of feat_1 is dead in the reference -> skipped
  for (int br = 0; br < 3; ++br) {
    const float* pts = branch_pts[br];
    k_knn<<<dim3(CEILDIV(NP1, 128), BATCH), 128, 0, stream>>>(pts, NP1, 3, 5, idxp, 4);
    long n = (long)BATCH * NP2 * 128;
    k_pool<<<CEILDIV(n, 256), 256, 0, stream>>>(fm1[br], idxp, fp1[br], NP1, NP2, 128, 4);
    long ns = (long)BATCH * NP2 * 3;
    k_slice<<<CEILDIV(ns, 256), 256, 0, stream>>>(pts, pp[br], NP1, NP2, 3);
  }
  {
    long n = (long)BATCH * NP2 * 9;
    k_slice<<<CEILDIV(n, 256), 256, 0, stream>>>(featfeat, p1, NP1, NP2, 9);
  }

  // ---- layer2
  k_knn<<<dim3(CEILDIV(NP2, 128), BATCH), 128, 0, stream>>>(pp[0], NP2, 3, KNN_K + 1, idx2, KNN_K);
  for (int br = 0; br < 3; ++br) {
    int base = 3 + 11 * br;
    gemm(fp1[br], P(base + 6), P(base + 7), fout, BATCH * NP2, 128, 1024);
    k_conv_act<<<BATCH * NP2, 128, 0, stream>>>(idx2, pp[br], dn2[br], fout, convout, NP2, 3, KNN_K, 128);
    bnrelu(convout, P(base + 9), P(base + 10), fm2[br], BATCH * NP2, 128);
  }
  {
    long n = (long)BATCH * NP2 * 384;
    k_concat3<<<CEILDIV(n, 256), 256, 0, stream>>>(fm2[0], fm2[1], fm2[2], feat2, NP2, 128);
  }

  // ---- level-2 pooling
  k_knn<<<dim3(CEILDIV(NP2, 128), BATCH), 128, 0, stream>>>(p1, NP2, 9, 5, idxp2, 4);
  {
    long n = (long)BATCH * NP3 * 384;
    k_pool<<<CEILDIV(n, 256), 256, 0, stream>>>(feat2, idxp2, fp2, NP2, NP3, 384, 4);
    long n9 = (long)BATCH * NP3 * 9;
    k_slice<<<CEILDIV(n9, 256), 256, 0, stream>>>(p1, p2, NP2, NP3, 9);
    long n3 = (long)BATCH * NP3 * 3;
    k_slice<<<CEILDIV(n3, 256), 256, 0, stream>>>(Vts, vp2, NP1, NP3, 3);
  }

  // ---- layer4 / layer5 (512-channel, 9-dim directions, no BN)
  k_knn<<<dim3(CEILDIV(NP3, 128), BATCH), 128, 0, stream>>>(p2, NP3, 9, KNN_K + 1, idx3, KNN_K);
  gemm(fp2, P(IW4), P(IB4), fout, BATCH * NP3, 384, 4096);
  k_conv_act<<<BATCH * NP3, 512, 0, stream>>>(idx3, p2, dn4, fout, fm4, NP3, 9, KNN_K, 512);
  gemm(fm4, P(IW5), P(IB5), fout, BATCH * NP3, 512, 4096);
  k_conv_act<<<BATCH * NP3, 512, 0, stream>>>(idx3, p2, dn5, fout, fm5, NP3, 9, KNN_K, 512);

  // ---- nearest-neighbor upsample indices + final concat
  k_nearest<<<dim3(CEILDIV(NP1, 128), BATCH), 128, 0, stream>>>(Vts, pp[0], NP1, NP2, 3, ni1);
  k_nearest<<<dim3(CEILDIV(NP1, 128), BATCH), 128, 0, stream>>>(Vts, vp2, NP1, NP3, 3, ni2);
  {
    long n = (long)BATCH * NP1 * 1280;
    k_final<<<CEILDIV(n, 256), 256, 0, stream>>>(fm5, feat1, feat2, ni1, ni2, (float*)d_out);
  }
}
